// Quantizer_16690242912270
// MI455X (gfx1250) — matrix operand order repaired and checked by reading, not compiled, not run on hardware
//
#include <hip/hip_runtime.h>

typedef _Float16 f16;
typedef _Float16 v8h  __attribute__((ext_vector_type(8)));
typedef _Float16 v16h __attribute__((ext_vector_type(16)));
typedef float    v8f  __attribute__((ext_vector_type(8)));

#define C_DIM 8192
#define D_DIM 512
#define B_DIM 4
#define P_DIM 1024                      // H*W
#define TAU_INV (1.0f / 0.9f)
#define LOG_UNIFORM (-9.0109133472f)    // log(1/8192)

// ---------------------------------------------------------------------------
// CDNA5 async copy: global -> LDS, 16B per lane, tracked by ASYNCcnt.
// ---------------------------------------------------------------------------
__device__ __forceinline__ void async_copy_b128(const f16* gsrc, f16* ldst) {
  unsigned lofs = (unsigned)(uintptr_t)ldst;              // LDS byte offset
  unsigned long long ga = (unsigned long long)(uintptr_t)gsrc;
  asm volatile("global_load_async_to_lds_b128 %0, %1, off"
               :: "v"(lofs), "v"(ga) : "memory");
}

// ---------------------------------------------------------------------------
// emb [C,D] f32  ->  embT [D,C] f16   (coalesced writes; strided reads hit L2)
// ---------------------------------------------------------------------------
__global__ __launch_bounds__(256) void k_embT(const float* __restrict__ emb,
                                              f16* __restrict__ embT) {
  int idx = blockIdx.x * 256 + threadIdx.x;     // D*C total
  int c = idx & (C_DIM - 1);
  int d = idx >> 13;
  embT[(size_t)d * C_DIM + c] = (f16)emb[(size_t)c * D_DIM + d];
}

// ---------------------------------------------------------------------------
// Per-pixel softmaxes. 16 pixels per block (consecutive w) => coalesced loads.
// Pass 1: online (max,sum) for z=(x+g)/tau and for x.  LDS merge across the
// 16 channel-slots per pixel.  Pass 2: tiled transpose-through-LDS writes of
// logits (=x), log_qy, one_hot(f16, transposed for the GEMM), KL partials.
// ---------------------------------------------------------------------------
__global__ __launch_bounds__(256) void k_softmax(const float* __restrict__ x,
                                                 const float* __restrict__ g,
                                                 f16* __restrict__ At,
                                                 float* __restrict__ logits,
                                                 float* __restrict__ logqy,
                                                 float* __restrict__ klpart) {
  const int t  = threadIdx.x;
  const int q  = t & 15;                 // pixel within block
  const int cs = t >> 4;                 // channel slot 0..15
  const int blk = blockIdx.x;            // 256 blocks
  const int b   = blk >> 6;              // 64 blocks per batch
  const int p0  = (blk & 63) << 4;
  const size_t xbase = ((size_t)b * C_DIM) * P_DIM + (size_t)(p0 + q);

  // ---- pass 1: online softmax statistics ---------------------------------
  float mz = -INFINITY, sz = 0.f, mx = -INFINITY, sx = 0.f;
#pragma unroll 4
  for (int i = 0; i < C_DIM / 16; ++i) {
    int c = cs + (i << 4);
    float xv = x[xbase + (size_t)c * P_DIM];
    float gv = g[xbase + (size_t)c * P_DIM];
    float zv = (xv + gv) * TAU_INV;
    float nm = fmaxf(mz, zv);
    sz = sz * __expf(mz - nm) + __expf(zv - nm);  mz = nm;
    nm = fmaxf(mx, xv);
    sx = sx * __expf(mx - nm) + __expf(xv - nm);  mx = nm;
  }

  __shared__ float r0[256], r1[256], r2[256], r3[256];
  r0[t] = mz; r1[t] = sz; r2[t] = mx; r3[t] = sx;
  __syncthreads();
  for (int off = 8; off > 0; off >>= 1) {
    if (cs < off) {
      int o = t + (off << 4);
      float m1 = r0[t], s1 = r1[t], m2 = r0[o], s2 = r1[o];
      float nm = fmaxf(m1, m2);
      r1[t] = s1 * __expf(m1 - nm) + s2 * __expf(m2 - nm);  r0[t] = nm;
      m1 = r2[t]; s1 = r3[t]; m2 = r2[o]; s2 = r3[o];
      nm = fmaxf(m1, m2);
      r3[t] = s1 * __expf(m1 - nm) + s2 * __expf(m2 - nm);  r2[t] = nm;
    }
    __syncthreads();
  }
  const float Mz    = r0[q];
  const float rSz   = 1.0f / r1[q];
  const float Mx    = r2[q];
  const float logSx = __logf(r3[q]);
  __syncthreads();

  // ---- pass 2: tiled outputs (transpose via LDS => coalesced writes) -----
  __shared__ float tl[16][65];
  __shared__ float tq[16][65];
  __shared__ f16   th[16][68];
  const int wr = t >> 4;                 // writer-phase pixel row
  const int wj = t & 15;                 // writer-phase column slot
  float kl = 0.f;

  for (int cb = 0; cb < C_DIM; cb += 64) {
#pragma unroll
    for (int u = 0; u < 4; ++u) {
      int cc = cs + (u << 4);
      int c  = cb + cc;
      float xv = x[xbase + (size_t)c * P_DIM];
      float gv = g[xbase + (size_t)c * P_DIM];
      float zv = (xv + gv) * TAU_INV;
      float oh = __expf(zv - Mz) * rSz;
      float lq = xv - Mx - logSx;
      kl += __expf(lq) * (lq - LOG_UNIFORM);
      tl[q][cc] = xv;
      tq[q][cc] = lq;
      th[q][cc] = (f16)oh;
    }
    __syncthreads();
    const size_t orow = ((size_t)b * P_DIM + (size_t)(p0 + wr)) * C_DIM + cb;
#pragma unroll
    for (int u = 0; u < 4; ++u) {
      int cc = (u << 4) + wj;
      logits[orow + cc] = tl[wr][cc];
      logqy [orow + cc] = tq[wr][cc];
      At    [orow + cc] = th[wr][cc];
    }
    __syncthreads();
  }

  // ---- KL partial per pixel (fixed-order tree => deterministic) ----------
  r0[t] = kl;
  __syncthreads();
  for (int off = 8; off > 0; off >>= 1) {
    if (cs < off) r0[t] += r0[t + (off << 4)];
    __syncthreads();
  }
  if (cs == 0) klpart[(blk << 4) + q] = r0[q];
}

// ---------------------------------------------------------------------------
// Deterministic final KL reduction: 4096 -> 1
// ---------------------------------------------------------------------------
__global__ __launch_bounds__(256) void k_klsum(const float* __restrict__ klpart,
                                               float* __restrict__ out) {
  __shared__ float s[256];
  int t = threadIdx.x;
  float a = 0.f;
  for (int i = t; i < B_DIM * P_DIM; i += 256) a += klpart[i];
  s[t] = a;
  __syncthreads();
  for (int off = 128; off > 0; off >>= 1) {
    if (t < off) s[t] += s[t + off];
    __syncthreads();
  }
  if (t == 0) out[0] = s[0];
}

// ---------------------------------------------------------------------------
// sampled[b,d,p] = sum_c embT[d,c] * one_hotT[b,p,c]  via WMMA f16 16x16x32.
//   A = embT (M=d, K=c) direct from global (unique rows per wave).
//   B = one_hotT (K=c, N=p) is shared by all 8 waves of the block, so it is
//   staged in LDS ONCE per block via global_load_async_to_lds_b128 with a
//   2-deep double buffer (ASYNCcnt), then read as ds_load_b128 fragments.
//   Row pitch 72 halves => 36-dword stride => 16 conflict-free banks.
//   D f32 fragment: lane = N = pixel => coalesced 64B stores into [B,D,H,W].
// ---------------------------------------------------------------------------
__global__ __launch_bounds__(256) void k_gemm(const f16* __restrict__ embT,
                                              const f16* __restrict__ At,
                                              float* __restrict__ sampled) {
  __shared__ f16 bsm[2][64 * 72];                // 2 x 9216 B double buffer
  const int t       = threadIdx.x;
  const int lane    = t & 31;
  const int wv      = t >> 5;                    // 0..7
  const int b       = blockIdx.z;
  const int d0      = blockIdx.x * 128 + wv * 16;
  const int pblk    = blockIdx.y * 64;
  const int halfSel = lane >> 4;                 // 0: lanes 0-15, 1: lanes 16-31
  const int mrow    = lane & 15;

  // A (embT) fragment: lanes 0-15 K in {0..7,16..23}; lanes 16-31 {8..15,24..31}
  const f16* aRow = embT + (size_t)(d0 + mrow) * C_DIM;
  const int  aKlo = halfSel ? 8 : 0;
  // B fragment: lanes 0-15 K in {0..7,16..23}; lanes 16-31 {8..15,24..31}
  const int  bKoff = halfSel ? 8 : 0;

  // async staging map: thread t moves 32B of the 64px x 64ch tile
  const int sp = t >> 2;                         // pixel row 0..63
  const int sc = (t & 3) << 4;                   // half offset 0,16,32,48
  const f16* gRow = At + ((size_t)b * P_DIM + (size_t)(pblk + sp)) * C_DIM + sc;
  f16* lRow0 = &bsm[0][sp * 72 + sc];
  f16* lRow1 = &bsm[1][sp * 72 + sc];

  // prologue: fill both buffers (4 async instructions in flight per wave)
  async_copy_b128(gRow +  0, lRow0);
  async_copy_b128(gRow +  8, lRow0 + 8);
  async_copy_b128(gRow + 64, lRow1);
  async_copy_b128(gRow + 72, lRow1 + 8);

  v8f acc[4] = {};
  for (int kb = 0; kb < C_DIM; kb += 64) {
    const int  cur = (kb >> 6) & 1;
    const f16* ls  = bsm[cur];
    __builtin_prefetch(aRow + kb + 512, 0, 1);
    // wait until the 2 oldest async ops (current buffer) have landed;
    // the 2 ops for the other buffer remain in flight.
    asm volatile("s_wait_asynccnt 0x2" ::: "memory");
    __syncthreads();
#pragma unroll
    for (int ks = 0; ks < 64; ks += 32) {
      v8h alo = *(const v8h*)(aRow + kb + ks + aKlo);
      v8h ahi = *(const v8h*)(aRow + kb + ks + aKlo + 16);
      v16h a;
#pragma unroll
      for (int i = 0; i < 8; ++i) { a[i] = alo[i]; a[8 + i] = ahi[i]; }
#pragma unroll
      for (int n = 0; n < 4; ++n) {
        const f16* lp = ls + (size_t)((n << 4) + mrow) * 72 + ks + bKoff;
        v8h blo = *(const v8h*)lp;
        v8h bhi = *(const v8h*)(lp + 16);
        v16h bf;
#pragma unroll
        for (int i = 0; i < 8; ++i) { bf[i] = blo[i]; bf[8 + i] = bhi[i]; }
        acc[n] = __builtin_amdgcn_wmma_f32_16x16x32_f16(
            false, a, false, bf, (short)0, acc[n], false, false);
      }
    }
    __syncthreads();   // all waves done reading buffer `cur` before refill
    {
      const int kn = (kb + 128) & (C_DIM - 1);   // tail wraps (harmless reload)
      f16* l = cur ? lRow1 : lRow0;
      async_copy_b128(gRow + kn,     l);
      async_copy_b128(gRow + kn + 8, l + 8);
    }
  }
  // D fragment: lane => N=pixel (contiguous across lanes 0-15 => 64B stores),
  // VGPR v => M=d (= d0 + 8*halfSel + v)
#pragma unroll
  for (int n = 0; n < 4; ++n) {
    const int p = pblk + (n << 4) + mrow;
#pragma unroll
    for (int v = 0; v < 8; ++v) {
      const int d = d0 + (halfSel << 3) + v;
      sampled[((size_t)b * D_DIM + d) * P_DIM + p] = acc[n][v];
    }
  }
}

// ---------------------------------------------------------------------------
extern "C" void kernel_launch(void* const* d_in, const int* in_sizes, int n_in,
                              void* d_out, int out_size, void* d_ws, size_t ws_size,
                              hipStream_t stream) {
  const float* x   = (const float*)d_in[0];   // [4,8192,32,32]
  const float* g   = (const float*)d_in[1];   // [4,8192,32,32]
  const float* emb = (const float*)d_in[2];   // [8192,512]

  float* out     = (float*)d_out;
  float* sampled = out;                                   // 2,097,152
  float* kl      = out + 2097152;                         // 1
  float* logits  = out + 2097153;                         // 33,554,432
  float* logqy   = out + 2097153 + 33554432;              // 33,554,432

  char* ws = (char*)d_ws;
  f16*   At     = (f16*)ws;                               // 67,108,864 B
  f16*   embT   = (f16*)(ws + 67108864);                  //  8,388,608 B
  float* klpart = (float*)(ws + 67108864 + 8388608);      //     16,384 B

  k_embT   <<<dim3((D_DIM * C_DIM) / 256), dim3(256), 0, stream>>>(emb, embT);
  k_softmax<<<dim3((B_DIM * P_DIM) / 16),  dim3(256), 0, stream>>>(x, g, At, logits, logqy, klpart);
  k_klsum  <<<dim3(1),                     dim3(256), 0, stream>>>(klpart, kl);
  k_gemm   <<<dim3(D_DIM / 128, P_DIM / 64, B_DIM), dim3(256), 0, stream>>>(embT, At, sampled);
}
